// RNade_65790309040624
// MI455X (gfx1250) — matrix operand order repaired and checked by reading, not compiled, not run on hardware
//
#include <hip/hip_runtime.h>

// RNADE fused kernel for MI455X (gfx1250, wave32, WMMA).
// B=2048, D=128, H=512, K=10. Output: (B,) float32.

typedef __attribute__((ext_vector_type(16))) _Float16 v16h;
typedef __attribute__((ext_vector_type(8)))  float    v8f;

#define Bsz 2048
#define Dd  128
#define Hh  512
#define Kk  10

// ---- dynamic LDS layout (bytes) ----
// x tile  : 16 rows x 132 floats (padded)   = 8448
// hA      : 4 dslots x 16 kchunks x 1024B   = 65536  (A fragments, f16)
// zbuf    : 4 dslots x 16 m x 32 n floats   = 8192
// biases  : 3 x (K*D) floats                = 15360
// red     : 64 floats                       = 256
#define XT_OFF    0
#define HA_OFF    8448
#define ZB_OFF    (8448 + 65536)
#define BIAS_OFF  (8448 + 65536 + 8192)
#define RED_OFF   (BIAS_OFF + 15360)
#define SMEM_BYTES (RED_OFF + 256)

// =====================================================================
// Kernel 1: pack V_alpha/V_mu/V_sigma (f32, (K,H,D), d innermost) into
// f16 B-matrix fragments for v_wmma_f32_16x16x32_f16.
// Fragment = one 32(K=H-chunk) x 16(N=output) tile, 1024 bytes:
//   dword index = lane*8 + v ; lane<16 -> Krows 0..15 (2v,2v+1),
//   lane>=16 -> Krows 16..31 ; N = lane&15.
// Output column n = mat*10 + k (mat: 0=alpha,1=mu,2=sigma); n>=30 zero.
// Vpack[frag=((d*2+tile)*16+kc)][256 dwords], total 4 MB in d_ws.
// =====================================================================
__global__ void __launch_bounds__(256)
rnade_pack_v(const float* __restrict__ Va, const float* __restrict__ Vm,
             const float* __restrict__ Vs, unsigned int* __restrict__ Vpack) {
  int frag = blockIdx.x;            // 0..4095
  int d    = frag >> 5;
  int tile = (frag >> 4) & 1;
  int kc   = frag & 15;
  int t    = threadIdx.x;           // 0..255
  int L = t >> 3, v = t & 7;
  int n  = tile * 16 + (L & 15);
  int kl = ((L >> 4) << 4) + (v << 1);     // (L>=16?16:0) + 2v
  int h0 = kc * 32 + kl;
  float f0 = 0.f, f1 = 0.f;
  if (n < 30) {
    int mat = n / 10, k = n % 10;
    const float* V = (mat == 0) ? Va : (mat == 1) ? Vm : Vs;
    f0 = V[((size_t)k * Hh + h0) * Dd + d];
    f1 = V[((size_t)k * Hh + h0 + 1) * Dd + d];
  }
  union { _Float16 h[2]; unsigned int u; } pk;
  pk.h[0] = (_Float16)f0;
  pk.h[1] = (_Float16)f1;
  Vpack[(size_t)frag * 256 + t] = pk.u;
}

// =====================================================================
// Kernel 2: fused NADE scan + per-d GEMM (WMMA) + mixture epilogue.
// One block = 16 batch rows, 256 threads = 8 wave32.
// Per thread: owns 32 cum slots (16 kchunks x 2 H-values) in registers,
// in this code's A-fragment k order (contiguous, as in Vpack):
//   A tile (16M x 32K): lane<16 M=0..15, v0..7 -> K=2v,2v+1;
//   lane>=16 same M, K += 16.
// =====================================================================
__global__ void __launch_bounds__(256)
rnade_main(const float* __restrict__ x, const float* __restrict__ W,
           const float* __restrict__ rho,
           const float* __restrict__ ba, const float* __restrict__ bm,
           const float* __restrict__ bs,
           const unsigned int* __restrict__ Vpack,
           float* __restrict__ out) {
  extern __shared__ char smem[];
  float* xt  = (float*)(smem + XT_OFF);   // [16][132]
  char*  hA  = smem + HA_OFF;             // [4][16][1024]
  float* zb  = (float*)(smem + ZB_OFF);   // [4*16][32]
  float* bia = (float*)(smem + BIAS_OFF); // alpha|mu|sigma, each K*D
  float* red = (float*)(smem + RED_OFF);  // [64]

  const int t    = threadIdx.x;
  const int lane = t & 31;
  const int wv   = t >> 5;
  const int bt   = blockIdx.x;

  // ---- preload x tile and biases into LDS
  for (int i = t; i < 16 * Dd; i += 256) {
    int m = i >> 7, dd = i & 127;
    xt[m * 132 + dd] = x[(size_t)(bt * 16 + m) * Dd + dd];
  }
  for (int i = t; i < 3 * Kk * Dd; i += 256) {
    float bvl = (i < Kk * Dd) ? ba[i]
              : (i < 2 * Kk * Dd) ? bm[i - Kk * Dd]
              : bs[i - 2 * Kk * Dd];
    bia[i] = bvl;
  }

  // ---- producer constants: this thread's (lane, vgpr-dword) in A layout
  const int pL  = t >> 3, pv = t & 7;
  const int pm  = pL & 15;                                      // batch row
  const int pkb = ((pL >> 4) << 4) + (pv << 1);

  float2 cum[16];
#pragma unroll
  for (int i = 0; i < 16; ++i) cum[i] = make_float2(0.f, 0.f);

  // ---- GEMM wave mapping / epilogue mapping
  const int gds = wv >> 1, gtile = wv & 1;   // dslot, output tile per wave
  const int eds = t >> 4,  em    = t & 15;   // epilogue (t<64)
  float acc = 0.f;

  __syncthreads();

  for (int db = 0; db < Dd / 4; ++db) {
    // ================= produce h (A fragments) for 4 d-steps ==========
#pragma unroll
    for (int dsl = 0; dsl < 4; ++dsl) {
      int d = db * 4 + dsl;
      float rhod = rho[d];                                  // uniform -> s_load
      float xsv  = (d == 0) ? 1.0f : xt[pm * 132 + (d - 1)];
      const float2* Wp = (const float2*)(W + (size_t)d * Hh);
      char* hAd = hA + dsl * 16384 + pL * 32 + pv * 4;
#pragma unroll
      for (int kc = 0; kc < 16; ++kc) {
        float2 wv2 = Wp[(kc * 32 + pkb) >> 1];              // global_load_b64
        cum[kc].x += wv2.x * xsv;
        cum[kc].y += wv2.y * xsv;
        float a0 = fmaxf(rhod * cum[kc].x, 0.f);
        float a1 = fmaxf(rhod * cum[kc].y, 0.f);
        union { _Float16 h[2]; unsigned int u; } pk;
        pk.h[0] = (_Float16)a0;
        pk.h[1] = (_Float16)a1;
        *(unsigned int*)(hAd + kc * 1024) = pk.u;           // ds_store_b32
      }
    }
    __syncthreads();

    // ================= per-wave GEMM: z(16x16) = h(16x512) x V(512x16) =
    {
      int d = db * 4 + gds;
      const char* vb = (const char*)Vpack +
                       ((size_t)(d * 32 + gtile * 16)) * 1024 + lane * 32;
      const char* ab = hA + gds * 16384 + lane * 32;
      v8f c = {0.f, 0.f, 0.f, 0.f, 0.f, 0.f, 0.f, 0.f};
#pragma unroll
      for (int kc = 0; kc < 16; ++kc) {
        v16h bfr = *(const v16h*)(vb + kc * 1024);          // global_load_b128 x2 (L2)
        v16h afr = *(const v16h*)(ab + kc * 1024);          // ds_load_b128 x2
        c = __builtin_amdgcn_wmma_f32_16x16x32_f16(
                false, afr, false, bfr, (short)0, c, false, false);
      }
      if (db + 1 < Dd / 4)
        __builtin_prefetch(vb + (size_t)4 * 32 * 1024, 0, 1); // global_prefetch_b8
      // C layout: lane<16 -> M=0..7 in c[0..7], lane>=16 -> M=8..15; N=lane&15
      int mb = (lane >> 4) * 8;
      int nn = gtile * 16 + (lane & 15);
#pragma unroll
      for (int r = 0; r < 8; ++r)
        zb[(gds * 16 + mb + r) * 32 + nn] = c[r];
    }
    __syncthreads();

    // ================= mixture-density epilogue per (dslot, m) ========
    if (t < 64) {
      int d = db * 4 + eds;
      const float* z = &zb[(eds * 16 + em) * 32];
      float xv = xt[em * 132 + d];
      float za[10], tk[10];
      float maxa = -1e30f;
#pragma unroll
      for (int k = 0; k < 10; ++k) {
        za[k] = z[k] + bia[k * Dd + d];
        maxa = fmaxf(maxa, za[k]);
      }
      float se = 0.f;
#pragma unroll
      for (int k = 0; k < 10; ++k) se += __expf(za[k] - maxa);
      float lsea = maxa + __logf(se);
      float maxt = -1e30f;
#pragma unroll
      for (int k = 0; k < 10; ++k) {
        float mu = z[10 + k] + bia[(Kk + k) * Dd + d];
        float ls = z[20 + k] + bia[(2 * Kk + k) * Dd + d];
        ls = fminf(fmaxf(ls, -20.f), 20.f);
        float df = xv - mu;
        tk[k] = -0.5f * __expf(-2.f * ls) * df * df + za[k] - ls;
        maxt = fmaxf(maxt, tk[k]);
      }
      float st = 0.f;
#pragma unroll
      for (int k = 0; k < 10; ++k) st += __expf(tk[k] - maxt);
      acc += maxt + __logf(st) - lsea - 0.9189385332f;  // 0.5*log(2*pi)
    }
    __syncthreads();
  }

  // ---- reduce the 4 dslot partials per batch row, store out (B,)
  if (t < 64) red[t] = acc;
  __syncthreads();
  if (t < 16)
    out[bt * 16 + t] = red[t] + red[16 + t] + red[32 + t] + red[48 + t];
}

// =====================================================================
extern "C" void kernel_launch(void* const* d_in, const int* in_sizes, int n_in,
                              void* d_out, int out_size, void* d_ws, size_t ws_size,
                              hipStream_t stream) {
  (void)in_sizes; (void)n_in; (void)out_size; (void)ws_size;
  const float* x   = (const float*)d_in[0];
  const float* W   = (const float*)d_in[1];
  const float* rho = (const float*)d_in[2];
  const float* Va  = (const float*)d_in[3];
  const float* Vm  = (const float*)d_in[4];
  const float* Vs  = (const float*)d_in[5];
  const float* ba  = (const float*)d_in[6];
  const float* bm  = (const float*)d_in[7];
  const float* bs  = (const float*)d_in[8];

  unsigned int* Vpack = (unsigned int*)d_ws;  // 4096 fragments * 1024 B = 4 MB

  // Pass 1: repack V into WMMA B-fragment layout (f16), once; read from L2 after.
  rnade_pack_v<<<Dd * 2 * 16, 256, 0, stream>>>(Va, Vm, Vs, Vpack);

  // Pass 2: fused scan + WMMA GEMM + mixture epilogue.
  rnade_main<<<Bsz / 16, 256, SMEM_BYTES, stream>>>(x, W, rho, ba, bm, bs,
                                                    Vpack, (float*)d_out);
}
